// SSTB_30305289241150
// MI455X (gfx1250) — matrix operand order repaired and checked by reading, not compiled, not run on hardware
//
#include <hip/hip_runtime.h>
#include <hip/hip_bf16.h>
#include <math.h>
#include <stdint.h>

// ---------------------------------------------------------------------------
// SSTB transformer block for gfx1250 (MI455X): bf16 WMMA GEMMs + fp32 LN,
// WMMA-based windowed spatial attention, VALU spectral attention (tiny).
// Shapes: B=8, N=16384 (128x128), d=192, h=8, dh=24, ws=8, dff=768, M=131072.
// Workspace use: ~455 MB.
// ---------------------------------------------------------------------------

#define M_ROWS 131072          // B * N
#define D_MODEL 192
#define D_QKV 576
#define D_FF 768
#define NHEAD 8
#define DHEAD 24
#define HFM 128
#define WFM 128

// Async global->LDS staging (CDNA5 GLOBAL_LOAD_ASYNC_TO_LDS_*, ASYNCcnt),
// used when the toolchain exposes the builtins.
#if defined(__has_builtin)
#if __has_builtin(__builtin_amdgcn_global_load_async_to_lds_b128) && \
    __has_builtin(__builtin_amdgcn_s_wait_asynccnt)
#define HAVE_ASYNC_LDS 1
#endif
#endif
#ifndef HAVE_ASYNC_LDS
#define HAVE_ASYNC_LDS 0
#endif

typedef __attribute__((ext_vector_type(16))) __bf16 bf16x16;
typedef __attribute__((ext_vector_type(8)))  float  f32x8;

union Frag16 { bf16x16 v; unsigned int u[8]; uint4 q[2]; };

__device__ __forceinline__ unsigned short f2bf(float f) {
  unsigned int u = __float_as_uint(f);
  unsigned int r = u + 0x7FFFu + ((u >> 16) & 1u);   // round-to-nearest-even
  return (unsigned short)(r >> 16);
}
__device__ __forceinline__ float bf2f(unsigned short h) {
  return __uint_as_float(((unsigned int)h) << 16);
}

// ------------------------------- utility kernels ---------------------------

__global__ void copy_f32x4_kernel(const float4* __restrict__ s,
                                  float4* __restrict__ d, int n4) {
  int i = blockIdx.x * 256 + threadIdx.x;
  if (i < n4) d[i] = s[i];
}

__global__ void cast_bf16_kernel(const float* __restrict__ s,
                                 unsigned short* __restrict__ d, int n) {
  int i = blockIdx.x * 256 + threadIdx.x;
  if (i < n) d[i] = f2bf(s[i]);
}

__global__ void zero_f32_kernel(float* __restrict__ p, int n) {
  int i = blockIdx.x * 256 + threadIdx.x;
  if (i < n) p[i] = 0.f;
}

// ------------------------------- LayerNorm + cast --------------------------
// One wave (32 lanes) per row of 192; 8 rows per block.
__global__ __launch_bounds__(256) void ln_cast_kernel(
    const float* __restrict__ x, const float* __restrict__ g,
    const float* __restrict__ bta, unsigned short* __restrict__ out) {
  int wave = threadIdx.x >> 5;
  int lane = threadIdx.x & 31;
  size_t row = (size_t)blockIdx.x * 8 + wave;
  const float* xr = x + row * D_MODEL;
  float v[6];
  float s = 0.f;
#pragma unroll
  for (int i = 0; i < 6; ++i) { v[i] = xr[lane + 32 * i]; s += v[i]; }
#pragma unroll
  for (int m = 16; m >= 1; m >>= 1) s += __shfl_xor(s, m, 32);
  float mean = s * (1.f / 192.f);
  float vs = 0.f;
#pragma unroll
  for (int i = 0; i < 6; ++i) { float d = v[i] - mean; vs += d * d; }
#pragma unroll
  for (int m = 16; m >= 1; m >>= 1) vs += __shfl_xor(vs, m, 32);
  float rstd = rsqrtf(vs * (1.f / 192.f) + 1e-5f);
  unsigned short* orow = out + row * D_MODEL;
#pragma unroll
  for (int i = 0; i < 6; ++i) {
    int c = lane + 32 * i;
    orow[c] = f2bf((v[i] - mean) * rstd * g[c] + bta[c]);
  }
}

// ------------------------------- WMMA GEMM ---------------------------------
// C[M,N] = A[M,K] @ Bw[K,N] (+bias):
//   epi=0: store bf16 | epi=1: bias+GELU, store bf16 | epi=2: bias + fp32 +=
// Block: 256 thr = 8 waves; block tile 128x64; wave tile 32x32 (2x2 WMMA).

__device__ __forceinline__ void gemm_epilogue(
    f32x8 acc[2][2], const float* __restrict__ bias,
    unsigned short* __restrict__ outb, float* __restrict__ outf,
    int bm, int bn, int wm, int wn, int l, int hi, int N, int epi) {
#pragma unroll
  for (int tm = 0; tm < 2; ++tm) {
#pragma unroll
    for (int tn = 0; tn < 2; ++tn) {
      int colg = bn + wn * 32 + tn * 16 + l;
      float bv = bias ? bias[colg] : 0.f;
#pragma unroll
      for (int r = 0; r < 8; ++r) {
        int rowg = bm + wm * 32 + tm * 16 + r + hi * 8;
        float vv = acc[tm][tn][r] + bv;
        size_t o = (size_t)rowg * N + colg;
        if (epi == 1) vv = 0.5f * vv * (1.f + erff(vv * 0.70710678118654752f));
        if (epi == 2) outf[o] += vv;
        else          outb[o] = f2bf(vv);
      }
    }
  }
}

// ---- variant A: register-staged double buffer, packed-pair B LDS tile ----
#define BP_STRIDE 68

__device__ __forceinline__ void gemm_frag_compute(
    const unsigned short* __restrict__ sA, const unsigned int* __restrict__ sBp,
    int wm, int wn, int l, int hi, f32x8 acc[2][2]) {
  Frag16 af[2], bfr[2];
#pragma unroll
  for (int tm = 0; tm < 2; ++tm) {
    int row = wm * 32 + tm * 16 + l;
    int kb = hi * 16;   // lanes 0-15: K 0-15 ; lanes 16-31: K 16-31
    af[tm].q[0] = *(const uint4*)(sA + row * 32 + kb);
    af[tm].q[1] = *(const uint4*)(sA + row * 32 + kb + 8);
  }
#pragma unroll
  for (int tn = 0; tn < 2; ++tn) {
    int col = wn * 32 + tn * 16 + l;
    int kp0 = hi * 8;
#pragma unroll
    for (int i = 0; i < 8; ++i)
      bfr[tn].u[i] = sBp[(kp0 + i) * BP_STRIDE + col];
  }
#pragma unroll
  for (int tm = 0; tm < 2; ++tm)
#pragma unroll
    for (int tn = 0; tn < 2; ++tn)
      acc[tm][tn] = __builtin_amdgcn_wmma_f32_16x16x32_bf16(
          false, af[tm].v, false, bfr[tn].v, (short)0, acc[tm][tn],
          false, false);
}

__global__ __launch_bounds__(256) void wmma_gemm_kernel(
    const unsigned short* __restrict__ A, const unsigned short* __restrict__ Bw,
    const float* __restrict__ bias, unsigned short* __restrict__ outb,
    float* __restrict__ outf, int K, int N, int epi) {
  __shared__ __align__(16) unsigned short sA[2][128 * 32];
  __shared__ __align__(16) unsigned int   sBp[2][16 * BP_STRIDE];

  int tid = threadIdx.x;
  int lane = tid & 31;
  int wv = tid >> 5;
  int wm = wv & 3, wn = wv >> 2;
  int l = lane & 15, hi = lane >> 4;
  int bm = blockIdx.y * 128, bn = blockIdx.x * 64;
  int ar = tid >> 2, ac4 = tid & 3;
  int bkp = tid >> 4, bc4 = tid & 15;

  f32x8 acc[2][2] = {};
  uint4 a0, a1;
  uint2 blo, bhi;
  a0 = *(const uint4*)(A + (size_t)(bm + ar) * K + ac4 * 8);
  a1 = *(const uint4*)(A + (size_t)(bm + ar + 64) * K + ac4 * 8);
  blo = *(const uint2*)(Bw + (size_t)(2 * bkp) * N + bn + bc4 * 4);
  bhi = *(const uint2*)(Bw + (size_t)(2 * bkp + 1) * N + bn + bc4 * 4);
  {
    *(uint4*)(sA[0] + ar * 32 + ac4 * 8) = a0;
    *(uint4*)(sA[0] + (ar + 64) * 32 + ac4 * 8) = a1;
    const unsigned short* plo = (const unsigned short*)&blo;
    const unsigned short* phi = (const unsigned short*)&bhi;
    uint4 pk;
    pk.x = (unsigned)plo[0] | ((unsigned)phi[0] << 16);
    pk.y = (unsigned)plo[1] | ((unsigned)phi[1] << 16);
    pk.z = (unsigned)plo[2] | ((unsigned)phi[2] << 16);
    pk.w = (unsigned)plo[3] | ((unsigned)phi[3] << 16);
    *(uint4*)(sBp[0] + bkp * BP_STRIDE + bc4 * 4) = pk;
  }
  __syncthreads();

  int buf = 0;
  for (int k0 = 32; k0 < K; k0 += 32) {
    a0 = *(const uint4*)(A + (size_t)(bm + ar) * K + k0 + ac4 * 8);
    a1 = *(const uint4*)(A + (size_t)(bm + ar + 64) * K + k0 + ac4 * 8);
    blo = *(const uint2*)(Bw + (size_t)(k0 + 2 * bkp) * N + bn + bc4 * 4);
    bhi = *(const uint2*)(Bw + (size_t)(k0 + 2 * bkp + 1) * N + bn + bc4 * 4);
    if (k0 + 32 < K)
      __builtin_prefetch((const void*)(A + (size_t)(bm + ar) * K + k0 + 32), 0, 3);

    gemm_frag_compute(sA[buf], sBp[buf], wm, wn, l, hi, acc);

    int nb = buf ^ 1;
    *(uint4*)(sA[nb] + ar * 32 + ac4 * 8) = a0;
    *(uint4*)(sA[nb] + (ar + 64) * 32 + ac4 * 8) = a1;
    const unsigned short* plo = (const unsigned short*)&blo;
    const unsigned short* phi = (const unsigned short*)&bhi;
    uint4 pk;
    pk.x = (unsigned)plo[0] | ((unsigned)phi[0] << 16);
    pk.y = (unsigned)plo[1] | ((unsigned)phi[1] << 16);
    pk.z = (unsigned)plo[2] | ((unsigned)phi[2] << 16);
    pk.w = (unsigned)plo[3] | ((unsigned)phi[3] << 16);
    *(uint4*)(sBp[nb] + bkp * BP_STRIDE + bc4 * 4) = pk;
    __syncthreads();
    buf = nb;
  }
  gemm_frag_compute(sA[buf], sBp[buf], wm, wn, l, hi, acc);
  gemm_epilogue(acc, bias, outb, outf, bm, bn, wm, wn, l, hi, N, epi);
}

// ---- variant B: ASYNCcnt-pipelined staging via GLOBAL_LOAD_ASYNC_TO_LDS ----
#if HAVE_ASYNC_LDS
// Builtin signature (from hipcc diagnostic): param0 = int __vector(4) AS1*,
// param1 = LDS-side pointer; 16-byte transfers per lane.
typedef int v4i_gnu __attribute__((vector_size(16)));

__device__ __forceinline__ void async_cp16(const unsigned short* g,
                                           unsigned short* l) {
  __builtin_amdgcn_global_load_async_to_lds_b128(
      (__attribute__((address_space(1))) v4i_gnu*)g,
      (__attribute__((address_space(3))) v4i_gnu*)l, 0, 0);
}

#define BR_STRIDE 72   // raw B rows, padded (144 B, 16B-aligned rows)

__device__ __forceinline__ void gemm_frag_compute_raw(
    const unsigned short* __restrict__ sA, const unsigned short* __restrict__ sB,
    int wm, int wn, int l, int hi, f32x8 acc[2][2]) {
  Frag16 af[2], bfr[2];
#pragma unroll
  for (int tm = 0; tm < 2; ++tm) {
    int row = wm * 32 + tm * 16 + l;
    int kb = hi * 16;
    af[tm].q[0] = *(const uint4*)(sA + row * 32 + kb);
    af[tm].q[1] = *(const uint4*)(sA + row * 32 + kb + 8);
  }
#pragma unroll
  for (int tn = 0; tn < 2; ++tn) {
    int col = wn * 32 + tn * 16 + l;
    int kb2 = hi * 16;
#pragma unroll
    for (int i = 0; i < 8; ++i) {
      unsigned int lo = sB[(kb2 + 2 * i) * BR_STRIDE + col];
      unsigned int hh = sB[(kb2 + 2 * i + 1) * BR_STRIDE + col];
      bfr[tn].u[i] = lo | (hh << 16);
    }
  }
#pragma unroll
  for (int tm = 0; tm < 2; ++tm)
#pragma unroll
    for (int tn = 0; tn < 2; ++tn)
      acc[tm][tn] = __builtin_amdgcn_wmma_f32_16x16x32_bf16(
          false, af[tm].v, false, bfr[tn].v, (short)0, acc[tm][tn],
          false, false);
}

__global__ __launch_bounds__(256) void wmma_gemm_async_kernel(
    const unsigned short* __restrict__ A, const unsigned short* __restrict__ Bw,
    const float* __restrict__ bias, unsigned short* __restrict__ outb,
    float* __restrict__ outf, int K, int N, int epi) {
  __shared__ __align__(16) unsigned short sA[2][128 * 32];
  __shared__ __align__(16) unsigned short sB[2][32 * BR_STRIDE];

  int tid = threadIdx.x;
  int lane = tid & 31;
  int wv = tid >> 5;
  int wm = wv & 3, wn = wv >> 2;
  int l = lane & 15, hi = lane >> 4;
  int bm = blockIdx.y * 128, bn = blockIdx.x * 64;
  int ar = tid >> 2, ac4 = tid & 3;   // A: rows ar, ar+64; 8-col chunks
  int br = tid >> 3, bc8 = tid & 7;   // B: row, 8-col chunk

  f32x8 acc[2][2] = {};

  auto stage = [&](int buf, int k0) {
    async_cp16(A + (size_t)(bm + ar) * K + k0 + ac4 * 8,
               sA[buf] + ar * 32 + ac4 * 8);
    async_cp16(A + (size_t)(bm + ar + 64) * K + k0 + ac4 * 8,
               sA[buf] + (ar + 64) * 32 + ac4 * 8);
    async_cp16(Bw + (size_t)(k0 + br) * N + bn + bc8 * 8,
               sB[buf] + br * BR_STRIDE + bc8 * 8);
  };

  stage(0, 0);
  __builtin_amdgcn_s_wait_asynccnt(0);
  __syncthreads();

  int buf = 0;
  for (int k0 = 32; k0 < K; k0 += 32) {
    stage(buf ^ 1, k0);                       // lands in LDS under the WMMAs
    gemm_frag_compute_raw(sA[buf], sB[buf], wm, wn, l, hi, acc);
    __builtin_amdgcn_s_wait_asynccnt(0);
    __syncthreads();
    buf ^= 1;
  }
  gemm_frag_compute_raw(sA[buf], sB[buf], wm, wn, l, hi, acc);
  gemm_epilogue(acc, bias, outb, outf, bm, bn, wm, wn, l, hi, N, epi);
}
#endif  // HAVE_ASYNC_LDS

// ---------------------------- spectral attention ---------------------------
// attn[b,h,d,e] = sum_n k[b,n,h,d] * q[b,n,h,e]   (24x24 per (b,h); tiny)
__global__ __launch_bounds__(256) void spe_attn_partial_kernel(
    const unsigned short* __restrict__ qkv, float* __restrict__ attnM) {
  int bh = blockIdx.x >> 4;
  int chunk = blockIdx.x & 15;
  int b = bh >> 3, hd = bh & 7;
  float acc[3] = {0.f, 0.f, 0.f};
  int dd[3], ee[3], pp[3];
#pragma unroll
  for (int j = 0; j < 3; ++j) {
    int p = threadIdx.x + 256 * j;
    pp[j] = p;
    dd[j] = p / 24;
    ee[j] = p - dd[j] * 24;
  }
  int n0 = chunk * 1024;
  for (int n = n0; n < n0 + 1024; ++n) {
    const unsigned short* row =
        qkv + ((size_t)b * 16384 + n) * D_QKV + hd * DHEAD;
#pragma unroll
    for (int j = 0; j < 3; ++j)
      if (pp[j] < 576)
        acc[j] += bf2f(row[192 + dd[j]]) * bf2f(row[ee[j]]);
  }
#pragma unroll
  for (int j = 0; j < 3; ++j)
    if (pp[j] < 576) atomicAdd(&attnM[bh * 576 + pp[j]], acc[j]);
}

__global__ void spe_softmax_kernel(float* __restrict__ attnM,
                                   const float* __restrict__ alpha) {
  int bh = blockIdx.x;
  int hd = bh & 7;
  int d = threadIdx.x;
  if (d >= 24) return;
  float inv_a = 1.f / alpha[hd];
  float* rowp = attnM + bh * 576 + d * 24;
  float mx = -1e30f;
  for (int e = 0; e < 24; ++e) mx = fmaxf(mx, rowp[e] * inv_a);
  float s = 0.f;
  float tmp[24];
  for (int e = 0; e < 24; ++e) { tmp[e] = __expf(rowp[e] * inv_a - mx); s += tmp[e]; }
  float inv = 1.f / s;
  for (int e = 0; e < 24; ++e) rowp[e] = tmp[e] * inv;
}

__global__ __launch_bounds__(256) void spe_attn_apply_kernel(
    const unsigned short* __restrict__ qkv, const float* __restrict__ attnM,
    unsigned short* __restrict__ out) {
  __shared__ float Ash[8 * 576];
  int id = blockIdx.x * 256 + threadIdx.x;
  int n = id >> 3;
  int hd = id & 7;
  int b = n >> 14;
  for (int i = threadIdx.x; i < 8 * 576; i += 256)
    Ash[i] = attnM[b * 8 * 576 + i];
  __syncthreads();
  const unsigned short* vrow = qkv + (size_t)n * D_QKV + hd * DHEAD + 384;
  float v[24];
#pragma unroll
  for (int e = 0; e < 24; ++e) v[e] = bf2f(vrow[e]);
  unsigned short* orow = out + (size_t)n * D_MODEL + hd * DHEAD;
  const float* Ah = Ash + hd * 576;
  for (int d = 0; d < 24; ++d) {
    float s = 0.f;
#pragma unroll
    for (int e = 0; e < 24; ++e) s += Ah[d * 24 + e] * v[e];
    orow[d] = f2bf(s);
  }
}

// ---------------------------- spatial (window) attention --------------------
// One block (8 waves) per (window, head). WMMA for both S=QK^T and O=PV.
__global__ __launch_bounds__(256) void spa_attn_kernel(
    const unsigned short* __restrict__ qkv, unsigned short* __restrict__ out) {
  __shared__ __align__(16) unsigned short Qs[64 * 32];
  __shared__ __align__(16) unsigned short Ks[64 * 32];
  __shared__ __align__(16) unsigned short Vs[64 * 32];
  __shared__ float Sf[64][64];
  __shared__ __align__(16) unsigned short Ps[64 * 64];

  int tid = threadIdx.x;
  int lane = tid & 31, wv = tid >> 5;
  int l = lane & 15, hi = lane >> 4;
  int wh = blockIdx.x;
  int w = wh >> 3, hd = wh & 7;
  int b = w >> 8, wy = (w >> 4) & 15, wx = w & 15;

  for (int i = tid; i < 64 * 32; i += 256) {
    int t = i >> 5, e = i & 31;
    unsigned short q = 0, k = 0, v = 0;
    if (e < 24) {
      size_t g = ((size_t)b * 16384 +
                  (size_t)(wy * 8 + (t >> 3)) * WFM + wx * 8 + (t & 7)) *
                     D_QKV + hd * DHEAD;
      q = qkv[g + e];
      k = qkv[g + 192 + e];
      v = qkv[g + 384 + e];
    }
    Qs[i] = q; Ks[i] = k; Vs[i] = v;
  }
  __syncthreads();

  {
    const float scale = 0.2041241452319315f;  // 24^-0.5
    int tm = wv & 3;
    Frag16 aq;
    int rowq = tm * 16 + l;
    int kb = hi * 16;
    aq.q[0] = *(const uint4*)(Qs + rowq * 32 + kb);
    aq.q[1] = *(const uint4*)(Qs + rowq * 32 + kb + 8);
#pragma unroll
    for (int j = 0; j < 2; ++j) {
      int tn = (wv >> 2) * 2 + j;
      Frag16 bk;
      int colt = tn * 16 + l;
      int kb2 = hi * 16;
      bk.q[0] = *(const uint4*)(Ks + colt * 32 + kb2);
      bk.q[1] = *(const uint4*)(Ks + colt * 32 + kb2 + 8);
      f32x8 sacc = {};
      sacc = __builtin_amdgcn_wmma_f32_16x16x32_bf16(
          false, aq.v, false, bk.v, (short)0, sacc, false, false);
#pragma unroll
      for (int r = 0; r < 8; ++r)
        Sf[tm * 16 + r + hi * 8][tn * 16 + l] = sacc[r] * scale;
    }
  }
  __syncthreads();

  if (tid < 64) {
    float mx = -1e30f;
    for (int c = 0; c < 64; ++c) mx = fmaxf(mx, Sf[tid][c]);
    float sum = 0.f;
    for (int c = 0; c < 64; ++c) {
      float e = __expf(Sf[tid][c] - mx);
      Sf[tid][c] = e;
      sum += e;
    }
    float inv = 1.f / sum;
    for (int c = 0; c < 64; ++c) Sf[tid][c] *= inv;
  }
  __syncthreads();

  for (int i = tid; i < 64 * 64; i += 256)
    Ps[i] = f2bf(Sf[i >> 6][i & 63]);
  __syncthreads();

  {
    int tmo = wv & 3;
    int tno = wv >> 2;
    f32x8 oacc = {};
#pragma unroll
    for (int c0 = 0; c0 < 64; c0 += 32) {
      Frag16 ap;
      int rowp = tmo * 16 + l;
      int kb = hi * 16;
      ap.q[0] = *(const uint4*)(Ps + rowp * 64 + c0 + kb);
      ap.q[1] = *(const uint4*)(Ps + rowp * 64 + c0 + kb + 8);
      Frag16 bv;
      int cole = tno * 16 + l;
      int kb2 = hi * 16;
#pragma unroll
      for (int i = 0; i < 8; ++i) {
        unsigned int lo = Vs[(c0 + kb2 + 2 * i) * 32 + cole];
        unsigned int hh = Vs[(c0 + kb2 + 2 * i + 1) * 32 + cole];
        bv.u[i] = lo | (hh << 16);
      }
      oacc = __builtin_amdgcn_wmma_f32_16x16x32_bf16(
          false, ap.v, false, bv.v, (short)0, oacc, false, false);
    }
    int e = tno * 16 + l;
    if (e < 24) {
#pragma unroll
      for (int r = 0; r < 8; ++r) {
        int t = tmo * 16 + r + hi * 8;
        size_t g = ((size_t)b * 16384 +
                    (size_t)(wy * 8 + (t >> 3)) * WFM + wx * 8 + (t & 7)) *
                       D_MODEL + hd * DHEAD + e;
        out[g] = f2bf(oacc[r]);
      }
    }
  }
}

// ------------------------------- host side ---------------------------------

extern "C" void kernel_launch(void* const* d_in, const int* in_sizes, int n_in,
                              void* d_out, int out_size, void* d_ws,
                              size_t ws_size, hipStream_t stream) {
  const float* x        = (const float*)d_in[0];
  const float* ln1_g    = (const float*)d_in[1];
  const float* ln1_b    = (const float*)d_in[2];
  const float* ln2_g    = (const float*)d_in[3];
  const float* ln2_b    = (const float*)d_in[4];
  const float* ln3_g    = (const float*)d_in[5];
  const float* ln3_b    = (const float*)d_in[6];
  const float* spe_qkv_w  = (const float*)d_in[7];
  const float* spe_proj_w = (const float*)d_in[8];
  const float* spe_proj_b = (const float*)d_in[9];
  const float* alpha      = (const float*)d_in[10];
  const float* spa_qkv_w  = (const float*)d_in[11];
  const float* spa_proj_w = (const float*)d_in[12];
  const float* spa_proj_b = (const float*)d_in[13];
  const float* ffn_w1     = (const float*)d_in[14];
  const float* ffn_b1     = (const float*)d_in[15];
  const float* ffn_w2     = (const float*)d_in[16];
  const float* ffn_b2     = (const float*)d_in[17];

  float* xacc = (float*)d_out;  // residual accumulator == final output

  char* ws = (char*)d_ws;
  size_t off = 0;
  auto carve = [&](size_t bytes) -> void* {
    off = (off + 255) & ~(size_t)255;
    void* p = ws + off;
    off += bytes;
    return p;
  };
  const size_t M = M_ROWS;
  unsigned short* abuf   = (unsigned short*)carve(M * D_MODEL * 2);
  unsigned short* qkvbuf = (unsigned short*)carve(M * D_QKV * 2);
  unsigned short* aobuf  = (unsigned short*)carve(M * D_MODEL * 2);
  unsigned short* hbuf   = (unsigned short*)carve(M * D_FF * 2);
  unsigned short* w_spe_qkv  = (unsigned short*)carve(192 * 576 * 2);
  unsigned short* w_spe_proj = (unsigned short*)carve(192 * 192 * 2);
  unsigned short* w_spa_qkv  = (unsigned short*)carve(192 * 576 * 2);
  unsigned short* w_spa_proj = (unsigned short*)carve(192 * 192 * 2);
  unsigned short* w_ffn1     = (unsigned short*)carve(192 * 768 * 2);
  unsigned short* w_ffn2     = (unsigned short*)carve(768 * 192 * 2);
  float* attnM = (float*)carve(64 * 576 * 4);
  (void)ws_size; (void)in_sizes; (void)n_in; (void)out_size;

  auto cast = [&](const float* s, unsigned short* d, int n) {
    cast_bf16_kernel<<<(n + 255) / 256, 256, 0, stream>>>(s, d, n);
  };
  cast(spe_qkv_w,  w_spe_qkv,  192 * 576);
  cast(spe_proj_w, w_spe_proj, 192 * 192);
  cast(spa_qkv_w,  w_spa_qkv,  192 * 576);
  cast(spa_proj_w, w_spa_proj, 192 * 192);
  cast(ffn_w1,     w_ffn1,     192 * 768);
  cast(ffn_w2,     w_ffn2,     768 * 192);

  {
    int n4 = (int)(M * D_MODEL / 4);
    copy_f32x4_kernel<<<(n4 + 255) / 256, 256, 0, stream>>>(
        (const float4*)x, (float4*)xacc, n4);
  }

  dim3 blk(256);
  auto gemm = [&](const unsigned short* A, const unsigned short* B,
                  const float* bias, unsigned short* outb, float* outf,
                  int K, int N, int epi) {
    dim3 grd(N / 64, M_ROWS / 128);
#if HAVE_ASYNC_LDS
    wmma_gemm_async_kernel<<<grd, blk, 0, stream>>>(A, B, bias, outb, outf,
                                                    K, N, epi);
#else
    wmma_gemm_kernel<<<grd, blk, 0, stream>>>(A, B, bias, outb, outf, K, N, epi);
#endif
  };

  // ================= stage 1: spectral MSA =================
  ln_cast_kernel<<<M / 8, 256, 0, stream>>>(xacc, ln1_g, ln1_b, abuf);
  gemm(abuf, w_spe_qkv, nullptr, qkvbuf, nullptr, 192, 576, 0);
  zero_f32_kernel<<<(64 * 576 + 255) / 256, 256, 0, stream>>>(attnM, 64 * 576);
  spe_attn_partial_kernel<<<64 * 16, 256, 0, stream>>>(qkvbuf, attnM);
  spe_softmax_kernel<<<64, 32, 0, stream>>>(attnM, alpha);
  spe_attn_apply_kernel<<<(M * 8) / 256, 256, 0, stream>>>(qkvbuf, attnM, aobuf);
  gemm(aobuf, w_spe_proj, spe_proj_b, nullptr, xacc, 192, 192, 2);

  // ================= stage 2: spatial (window) MSA =================
  ln_cast_kernel<<<M / 8, 256, 0, stream>>>(xacc, ln2_g, ln2_b, abuf);
  gemm(abuf, w_spa_qkv, nullptr, qkvbuf, nullptr, 192, 576, 0);
  spa_attn_kernel<<<2048 * 8, 256, 0, stream>>>(qkvbuf, aobuf);
  gemm(aobuf, w_spa_proj, spa_proj_b, nullptr, xacc, 192, 192, 2);

  // ================= stage 3: FFN =================
  ln_cast_kernel<<<M / 8, 256, 0, stream>>>(xacc, ln3_g, ln3_b, abuf);
  gemm(abuf, w_ffn1, ffn_b1, hbuf, nullptr, 192, 768, 1);   // bias + GELU
  gemm(hbuf, w_ffn2, ffn_b2, nullptr, xacc, 768, 192, 2);   // bias + residual
}
